// MonotonicNN_38929583571109
// MI455X (gfx1250) — matrix operand order repaired and checked by reading, not compiled, not run on hardware
//
#include <hip/hip_runtime.h>
#include <hip/hip_bf16.h>

typedef __attribute__((ext_vector_type(16))) _Float16 v16h;
typedef __attribute__((ext_vector_type(8)))  _Float16 v8h;
typedef __attribute__((ext_vector_type(8)))  float    v8f;

#define NB_STEPS 50
#define BATCH    8192
#define PI_F     3.14159265358979323846f
#define FLT_BIG  3.402823466e38f

// ---- workspace layout (bytes) ----
#define CCW_OFF    0          // 64 f32
#define STEPS_OFF  256        // 64 f32
#define IW0_OFF    512        // 16 A-frags: 16x32 (K zero-padded from 8)
#define IW1_OFF    16896      // 256x256 f16 packed A-frags (W^T tiles)
#define IW2_OFF    147968     // 256x256 f16 packed
#define IW3_OFF    279040     // 16x256 f16 packed (padded from 1x256 = iW3^T)
#define HW0_OFF    287232     // 16 A-frags (K zero-padded from 7)
#define HW1_OFF    303616     // 256x256 f16 packed
#define HW2_OFF    434688     // 256x256 f16 packed
#define HW3_OFF    565760     // 16x256 f16 packed (padded from 2x256 = hW3^T)
#define INTEG_OFF  573952     // BATCH f32: per-item quadrature sums

// ------------- WMMA fragment maps (16-bit, wave32; contiguous K order of this code) -------------
// Weights are the A matrix (rows = output features), activations are B.
// A (16x32): lane<16: halves i -> K = i (0..15); lane>=16: +16
// B (32x16): lane holds column (batch row) = lane%16; halves i -> K = i (+16, lane>=16)
// C/D: lane -> batch row = lane%16; reg r -> feature = r + 8*(lane>=16)
__device__ __forceinline__ int ka_map(int L, int i) {
  return i + ((L >= 16) ? 16 : 0);
}

__device__ __forceinline__ void pack256(const float* __restrict__ src,
                                        _Float16* __restrict__ dst, int idx) {
  int f  = idx >> 9;          // fragment: nt*8 + kt
  int w  = idx & 511;
  int L  = w >> 4, i = w & 15;
  int nt = f >> 3, kt = f & 7;
  int K  = kt * 32 + ka_map(L, i);
  int N  = nt * 16 + (L & 15);
  dst[idx] = (_Float16)src[K * 256 + N];
}

__device__ __forceinline__ void packout(const float* __restrict__ src,
                                        _Float16* __restrict__ dst, int idx, int ncols) {
  int kt = idx >> 9;
  int w  = idx & 511;
  int L  = w >> 4, i = w & 15;
  int K  = kt * 32 + ka_map(L, i);
  int Nl = L & 15;
  dst[idx] = (Nl < ncols) ? (_Float16)src[K * ncols + Nl] : (_Float16)0.f;
}

__device__ __forceinline__ void pack_in(const float* __restrict__ src,
                                        _Float16* __restrict__ dst, int idx, int krows) {
  int nt = idx >> 9;
  int w  = idx & 511;
  int L  = w >> 4, i = w & 15;
  int K  = ka_map(L, i);
  int Nl = L & 15;
  dst[idx] = (K < krows) ? (_Float16)src[K * 256 + nt * 16 + Nl] : (_Float16)0.f;
}

__global__ void __launch_bounds__(256)
prep_kernel(const float* __restrict__ iW0, const float* __restrict__ iW1,
            const float* __restrict__ iW2, const float* __restrict__ iW3,
            const float* __restrict__ hW0, const float* __restrict__ hW1,
            const float* __restrict__ hW2, const float* __restrict__ hW3,
            char* __restrict__ ws) {
  int id = blockIdx.x * 256 + threadIdx.x;
  if (id < 65536) { pack256(iW1, (_Float16*)(ws + IW1_OFF), id); return; }
  id -= 65536;
  if (id < 65536) { pack256(iW2, (_Float16*)(ws + IW2_OFF), id); return; }
  id -= 65536;
  if (id < 65536) { pack256(hW1, (_Float16*)(ws + HW1_OFF), id); return; }
  id -= 65536;
  if (id < 65536) { pack256(hW2, (_Float16*)(ws + HW2_OFF), id); return; }
  id -= 65536;
  if (id < 4096) { packout(iW3, (_Float16*)(ws + IW3_OFF), id, 1); return; }
  id -= 4096;
  if (id < 4096) { packout(hW3, (_Float16*)(ws + HW3_OFF), id, 2); return; }
  id -= 4096;
  if (id < 8192) { pack_in(iW0, (_Float16*)(ws + IW0_OFF), id, 8); return; }
  id -= 8192;
  if (id < 8192) { pack_in(hW0, (_Float16*)(ws + HW0_OFF), id, 7); return; }
  id -= 8192;
  if (id < 64) {   // Clenshaw-Curtis weights, padded to 64 with zeros
    float v = 0.f;
    if (id <= NB_STEPS) {
      float sum = 0.f;
      for (int i = 0; i <= NB_STEPS; ++i) {
        float Wq = (i == 0) ? 1.f : ((i & 1) ? 0.f : 2.f / (1.f - (float)(i * i)));
        float base;
        if (id == 0)             base = 0.5f;
        else if (id == NB_STEPS) base = 0.5f * __cosf((float)i * PI_F);
        else                     base = __cosf((float)(i * id) * PI_F / (float)NB_STEPS);
        sum += base * (2.f / (float)NB_STEPS) * Wq;
      }
      v = sum;
    }
    ((float*)(ws + CCW_OFF))[id] = v;
    return;
  }
  id -= 64;
  if (id < 64) {
    float v = (id <= NB_STEPS) ? __cosf((float)id * PI_F / (float)NB_STEPS) : 0.f;
    ((float*)(ws + STEPS_OFF))[id] = v;
  }
}

// ---------------- shared memory: 128-row tile ----------------
struct __align__(16) Smem {
  _Float16 bufA[128 * 256];   //  64 KB activations ping (row-major [m][k])
  _Float16 bufB[128 * 256];   //  64 KB activations pong
  float    outb[128 * 2];     //   1 KB hnet outputs
  float    red[128];          // 512 B reduction scratch
};

__device__ __forceinline__ v16h load16h(const _Float16* p) {
  v8h lo = *(const v8h*)p;
  v8h hi = *(const v8h*)(p + 8);
  v16h v;
#pragma unroll
  for (int i = 0; i < 8; ++i) { v[i] = lo[i]; v[i + 8] = hi[i]; }
  return v;
}

// ReLU via single v_med3_f32 + f16 pack + one contiguous 16B store
__device__ __forceinline__ void store_relu8(_Float16* p, const v8f& c) {
  v8h o;
#pragma unroll
  for (int r = 0; r < 8; ++r)
    o[r] = (_Float16)__builtin_amdgcn_fmed3f(c[r], 0.f, FLT_BIG);
  *(v8h*)p = o;
}

// layer 0 via WMMA, K zero-padded; B (z) frags prebuilt in registers.
__device__ __forceinline__ void layer0_blk(const v16h bz[4], _Float16* __restrict__ out,
                                           const _Float16* __restrict__ wp,
                                           const float* __restrict__ bias,
                                           int lane, int wave) {
  const int Ml = lane & 15, hi = lane >> 4;
  const int wm = wave >> 2, wn = wave & 3;
  const int m0 = wm * 64;
  v8f c[4][4];
#pragma unroll
  for (int t = 0; t < 4; ++t) {
    const float* bp = bias + (wn * 4 + t) * 16 + hi * 8;
#pragma unroll
    for (int j = 0; j < 4; ++j)
#pragma unroll
      for (int r = 0; r < 8; ++r) c[j][t][r] = bp[r];
  }
#pragma unroll
  for (int t = 0; t < 4; ++t) {
    v16h a = load16h(wp + (wn * 4 + t) * 512 + lane * 16);
#pragma unroll
    for (int j = 0; j < 4; ++j)
      c[j][t] = __builtin_amdgcn_wmma_f32_16x16x32_f16(
          false, a, false, bz[j], (short)0, c[j][t], false, false);
  }
#pragma unroll
  for (int t = 0; t < 4; ++t) {
    const int n0 = (wn * 4 + t) * 16 + hi * 8;
#pragma unroll
    for (int j = 0; j < 4; ++j)
      store_relu8(out + (m0 + j * 16 + Ml) * 256 + n0, c[j][t]);
  }
}

// 128x256 @ 256x256 layer: D = W^T-tile (A) x act-tile (B)
__device__ __forceinline__ void layer256_blk(const _Float16* __restrict__ in,
                                             _Float16* __restrict__ out,
                                             const _Float16* __restrict__ wp,
                                             const float* __restrict__ bias,
                                             int lane, int wave) {
  const int Ml = lane & 15, hi = lane >> 4;
  const int wm = wave >> 2, wn = wave & 3;
  const int m0 = wm * 64;
  v8f c[4][4];
#pragma unroll
  for (int t = 0; t < 4; ++t) {
    const float* bp = bias + (wn * 4 + t) * 16 + hi * 8;
#pragma unroll
    for (int j = 0; j < 4; ++j)
#pragma unroll
      for (int r = 0; r < 8; ++r) c[j][t][r] = bp[r];
  }
#pragma unroll
  for (int kt = 0; kt < 8; ++kt) {
    v16h bfr[4];
#pragma unroll
    for (int j = 0; j < 4; ++j)
      bfr[j] = load16h(in + (m0 + j * 16 + Ml) * 256 + kt * 32 + hi * 16);
#pragma unroll
    for (int t = 0; t < 4; ++t) {
      v16h a = load16h(wp + ((wn * 4 + t) * 8 + kt) * 512 + lane * 16);
#pragma unroll
      for (int j = 0; j < 4; ++j)
        c[j][t] = __builtin_amdgcn_wmma_f32_16x16x32_f16(
            false, a, false, bfr[j], (short)0, c[j][t], false, false);
    }
  }
#pragma unroll
  for (int t = 0; t < 4; ++t) {
    const int n0 = (wn * 4 + t) * 16 + hi * 8;
#pragma unroll
    for (int j = 0; j < 4; ++j)
      store_relu8(out + (m0 + j * 16 + Ml) * 256 + n0, c[j][t]);
  }
}

// final layer: one 16-row act tile per wave; lane<16 gets features 0..7 of its row
__device__ __forceinline__ v8f layer_out_blk(const _Float16* __restrict__ in,
                                             const _Float16* __restrict__ wp,
                                             v8f c, int lane, int wave) {
  const int Ml = lane & 15, hi = lane >> 4;
#pragma unroll
  for (int kt = 0; kt < 8; ++kt) {
    v16h a = load16h(wp + kt * 512 + lane * 16);
    v16h b = load16h(in + (wave * 16 + Ml) * 256 + kt * 32 + hi * 16);
    c = __builtin_amdgcn_wmma_f32_16x16x32_f16(false, a, false, b, (short)0, c, false, false);
  }
  return c;
}

// -------- integrand MLP + quadrature: one WG = 2 batch items x 64 steps ------
__global__ void __launch_bounds__(256)
integrand_kernel(const float* __restrict__ x, const float* __restrict__ h,
                 const float* __restrict__ ib0, const float* __restrict__ ib1,
                 const float* __restrict__ ib2, const float* __restrict__ ib3,
                 const char* __restrict__ ws, float* __restrict__ integ) {
  __shared__ Smem sm;
  const int tid  = threadIdx.x;
  const int lane = tid & 31, wave = tid >> 5;
  const int b0   = blockIdx.x * 2;

  const _Float16* w0 = (const _Float16*)(ws + IW0_OFF);
  const _Float16* w1 = (const _Float16*)(ws + IW1_OFF);
  const _Float16* w2 = (const _Float16*)(ws + IW2_OFF);
  const _Float16* w3 = (const _Float16*)(ws + IW3_OFF);
  const float* ccw   = (const float*)(ws + CCW_OFF);
  const float* steps = (const float*)(ws + STEPS_OFF);

  // layer-0 B fragments (z) in registers: lane<16 carries K=0..7 of its row
  v16h bz[4];
  {
    const int Ml = lane & 15;
    const int m0 = (wave >> 2) * 64;
#pragma unroll
    for (int j = 0; j < 4; ++j) {
      v16h v;
#pragma unroll
      for (int i = 0; i < 16; ++i) v[i] = (_Float16)0.f;
      if (lane < 16) {
        int row = m0 + j * 16 + Ml;
        int b   = b0 + (row >> 6);
        int s   = row & 63;
        v[0] = (_Float16)(x[b] * (steps[s] + 1.f) * 0.5f);
#pragma unroll
        for (int k = 1; k < 8; ++k) v[k] = (_Float16)h[b * 7 + (k - 1)];
      }
      bz[j] = v;
    }
  }
  layer0_blk(bz, sm.bufA, w0, ib0, lane, wave);
  __syncthreads();
  layer256_blk(sm.bufA, sm.bufB, w1, ib1, lane, wave);
  __syncthreads();
  layer256_blk(sm.bufB, sm.bufA, w2, ib2, lane, wave);
  __syncthreads();

  {
    v8f ci;
#pragma unroll
    for (int r = 0; r < 8; ++r) ci[r] = 0.f;
    if ((lane >> 4) == 0) ci[0] = ib3[0];
    v8f c = layer_out_blk(sm.bufA, w3, ci, lane, wave);
    if (lane < 16) {                     // feature 0 for row wave*16+lane
      int row = wave * 16 + lane;
      int s   = row & 63;
      float v  = c[0];
      float dv = (v > 0.f) ? (v + 1.f) : __expf(v);   // elu(v)+1
      sm.red[row] = dv * ccw[s];                      // padded ccw==0
    }
  }
  __syncthreads();
  if (tid < 2) {                         // finish quadrature sum per batch item
    const float* p = sm.red + tid * 64;
    float s = 0.f;
#pragma unroll
    for (int k = 0; k < 64; ++k) s += p[k];
    integ[b0 + tid] = s;
  }
}

// -------- h-network + final combine: one WG = 128 batch items ---------------
__global__ void __launch_bounds__(256)
hnet_kernel(const float* __restrict__ x, const float* __restrict__ h,
            const float* __restrict__ hb0, const float* __restrict__ hb1,
            const float* __restrict__ hb2, const float* __restrict__ hb3,
            const char* __restrict__ ws, float* __restrict__ out) {
  __shared__ Smem sm;
  const int tid  = threadIdx.x;
  const int lane = tid & 31, wave = tid >> 5;
  const int b0   = blockIdx.x * 128;

  const _Float16* w0 = (const _Float16*)(ws + HW0_OFF);
  const _Float16* w1 = (const _Float16*)(ws + HW1_OFF);
  const _Float16* w2 = (const _Float16*)(ws + HW2_OFF);
  const _Float16* w3 = (const _Float16*)(ws + HW3_OFF);
  const float* integ = (const float*)(ws + INTEG_OFF);

  // layer-0 B fragments: z = [h(7), 0]
  v16h bz[4];
  {
    const int Ml = lane & 15;
    const int m0 = (wave >> 2) * 64;
#pragma unroll
    for (int j = 0; j < 4; ++j) {
      v16h v;
#pragma unroll
      for (int i = 0; i < 16; ++i) v[i] = (_Float16)0.f;
      if (lane < 16) {
        int b = b0 + m0 + j * 16 + Ml;
#pragma unroll
        for (int k = 0; k < 7; ++k) v[k] = (_Float16)h[b * 7 + k];
      }
      bz[j] = v;
    }
  }
  layer0_blk(bz, sm.bufA, w0, hb0, lane, wave);
  __syncthreads();
  layer256_blk(sm.bufA, sm.bufB, w1, hb1, lane, wave);
  __syncthreads();
  layer256_blk(sm.bufB, sm.bufA, w2, hb2, lane, wave);
  __syncthreads();

  {
    v8f ci;
#pragma unroll
    for (int r = 0; r < 8; ++r) ci[r] = 0.f;
    if ((lane >> 4) == 0) { ci[0] = hb3[0]; ci[1] = hb3[1]; }
    v8f c = layer_out_blk(sm.bufA, w3, ci, lane, wave);
    if (lane < 16) {
      int row = wave * 16 + lane;
      sm.outb[row * 2 + 0] = c[0];
      sm.outb[row * 2 + 1] = c[1];
    }
  }
  __syncthreads();
  if (tid < 128) {
    int b = b0 + tid;
    float integral = 0.5f * x[b] * integ[b];
    out[b] = __expf(sm.outb[tid * 2 + 1]) * integral + sm.outb[tid * 2];
  }
}

extern "C" void kernel_launch(void* const* d_in, const int* in_sizes, int n_in,
                              void* d_out, int out_size, void* d_ws, size_t ws_size,
                              hipStream_t stream) {
  (void)in_sizes; (void)n_in; (void)out_size; (void)ws_size;
  const float* x   = (const float*)d_in[0];
  const float* h   = (const float*)d_in[1];
  const float* iW0 = (const float*)d_in[2];
  const float* ib0 = (const float*)d_in[3];
  const float* iW1 = (const float*)d_in[4];
  const float* ib1 = (const float*)d_in[5];
  const float* iW2 = (const float*)d_in[6];
  const float* ib2 = (const float*)d_in[7];
  const float* iW3 = (const float*)d_in[8];
  const float* ib3 = (const float*)d_in[9];
  const float* hW0 = (const float*)d_in[10];
  const float* hb0 = (const float*)d_in[11];
  const float* hW1 = (const float*)d_in[12];
  const float* hb1 = (const float*)d_in[13];
  const float* hW2 = (const float*)d_in[14];
  const float* hb2 = (const float*)d_in[15];
  const float* hW3 = (const float*)d_in[16];
  const float* hb3 = (const float*)d_in[17];
  char* ws = (char*)d_ws;
  float* integ = (float*)(ws + INTEG_OFF);

  const int prep_work = 4 * 65536 + 2 * 4096 + 2 * 8192 + 2 * 64;  // 286848
  prep_kernel<<<(prep_work + 255) / 256, 256, 0, stream>>>(iW0, iW1, iW2, iW3,
                                                           hW0, hW1, hW2, hW3, ws);
  integrand_kernel<<<BATCH / 2, 256, 0, stream>>>(x, h, ib0, ib1, ib2, ib3, ws, integ);
  hnet_kernel<<<BATCH / 128, 256, 0, stream>>>(x, h, hb0, hb1, hb2, hb3, ws,
                                               (float*)d_out);
}
